// PairUpdate_70660801953926
// MI455X (gfx1250) — matrix operand order repaired and checked by reading, not compiled, not run on hardware
//
#include <hip/hip_runtime.h>

typedef __attribute__((ext_vector_type(16))) _Float16 v16h;
typedef __attribute__((ext_vector_type(8)))  float    v8f;

#define N_SEQ 512
#define C_DIM 128
#define NPOS  (N_SEQ * N_SEQ)   // 262144 positions

union Frag16 { v16h h; uint4 u[2]; };

__device__ __forceinline__ float sigf(float x) { return 1.0f / (1.0f + __expf(-x)); }

// Load a 16-half fragment as two 16-byte chunks (p0: halves 0..7, p1: halves 8..15)
__device__ __forceinline__ v16h frag_from(const _Float16* p0, const _Float16* p1) {
  Frag16 f;
  f.u[0] = *(const uint4*)p0;
  f.u[1] = *(const uint4*)p1;
  return f.h;
}

__device__ __forceinline__ v8f wmma_f16(v16h a, v16h b, v8f c) {
  return __builtin_amdgcn_wmma_f32_16x16x32_f16(false, a, false, b, (short)0, c,
                                                false, false);
}

// CDNA5 async memory->LDS copy, 16 bytes per lane, tracked by ASYNCcnt.
// lds_off: LDS byte offset (low 32 bits of a generic LDS pointer == LDS offset).
__device__ __forceinline__ void async_b128(unsigned lds_off, const void* g) {
  asm volatile("global_load_async_to_lds_b128 %0, %1, off"
               :: "v"(lds_off), "v"((unsigned long long)(uintptr_t)g)
               : "memory");
}
#define WAIT_ASYNC(n) asm volatile("s_wait_asynccnt " #n ::: "memory")

__device__ __forceinline__ unsigned lds_off_of(const void* p) {
  return (unsigned)(uintptr_t)p;
}

// ---------------------------------------------------------------------------
// Kernel 0: swizzle six 128x128 f32 weights into B-fragment order, f16.
// Fragment f = kt*8 + nt (kt: K/32, nt: N/16); within fragment: lane L holds
// for column n = nt*16 + L%16 the K values kt*32 + 8*(L>=16) + {0..7, 16..23}.
// ---------------------------------------------------------------------------
__global__ void prep_weights(const float* wq1, const float* wq2, const float* wk1,
                             const float* wk2, const float* wv2, const float* wv1,
                             _Float16* wsW) {
  int idx = blockIdx.x * blockDim.x + threadIdx.x;   // one output half each
  const int per = C_DIM * C_DIM;                     // 16384
  if (idx >= 6 * per) return;
  int proj = idx / per;
  int r    = idx % per;
  int frag = r / 512;             // 32 lanes * 16 halves
  int rem  = r % 512;
  int lane = rem / 16;
  int h    = rem % 16;
  int kt = frag >> 3, nt = frag & 7;
  int n = nt * 16 + (lane & 15);
  int k = kt * 32 + ((lane < 16) ? 0 : 8) + ((h < 8) ? h : h + 8);
  const float* w = (proj == 0) ? wq1 : (proj == 1) ? wq2 : (proj == 2) ? wk1
                 : (proj == 3) ? wk2 : (proj == 4) ? wv2 : wv1;
  wsW[idx] = (_Float16)w[k * C_DIM + n];
}

// ---------------------------------------------------------------------------
// Kernel 1: LN(z) + 5 projections (q1,q2,k1,k2,v2), gating, mask.
//   Q -> wsQ[c][i][j] f16, K -> wsK[c][i][j] f16, V -> wsV[pos][c] f16
// One block = 16 positions x 128 channels; 8 waves; 20 WMMAs / wave.
// ---------------------------------------------------------------------------
__global__ __launch_bounds__(256)
void pass_a(const float* __restrict__ z, const float* __restrict__ mask,
            const float* __restrict__ bq1, const float* __restrict__ bq2,
            const float* __restrict__ bk1, const float* __restrict__ bk2,
            const float* __restrict__ bv2, const float* __restrict__ g_in,
            const float* __restrict__ b_in, const _Float16* __restrict__ wsW,
            _Float16* __restrict__ wsQ, _Float16* __restrict__ wsK,
            _Float16* __restrict__ wsV) {
  __shared__ __align__(16) float    zrow[16][C_DIM];   // 8 KB
  __shared__ float    red1[16][16], red2[16][16];
  __shared__ float    smask[16];
  __shared__ __align__(16) _Float16 zn[16][C_DIM];     // 4 KB

  const int tid = threadIdx.x;
  const int posbase = blockIdx.x * 16;

  // async stage the 16x128 f32 z tile: 32 B per thread
  {
    const float* gz = z + (size_t)posbase * C_DIM + tid * 8;
    unsigned l0 = lds_off_of(&zrow[0][0] + tid * 8);
    async_b128(l0,      gz);
    async_b128(l0 + 16, gz + 4);
  }
  if (tid < 16) smask[tid] = mask[posbase + tid];
  WAIT_ASYNC(0);
  __syncthreads();

  // LayerNorm: 16 threads per row, 8 elements each
  {
    int row = tid >> 4, l = tid & 15;
    float s1 = 0.f, s2 = 0.f;
    for (int e = 0; e < 8; ++e) {
      float x = zrow[row][l * 8 + e];
      s1 += x; s2 += x * x;
    }
    red1[row][l] = s1; red2[row][l] = s2;
  }
  __syncthreads();
  {
    int row = tid >> 4, l = tid & 15;
    float s1 = 0.f, s2 = 0.f;
    for (int e = 0; e < 16; ++e) { s1 += red1[row][e]; s2 += red2[row][e]; }
    float mu  = s1 * (1.f / 128.f);
    float var = s2 * (1.f / 128.f) - mu * mu;
    float inv = rsqrtf(var + 1e-5f);
    for (int e = 0; e < 8; ++e) {
      int c = l * 8 + e;
      zn[row][c] = (_Float16)((zrow[row][c] - mu) * inv * g_in[c] + b_in[c]);
    }
  }
  __syncthreads();

  const int wid = tid >> 5, lane = tid & 31;
  const int hl = lane & 15, hi = lane >> 4;

  // A fragments: row M = hl; lanes 0-15 take K {0..7,16..23}, 16-31 {8..15,24..31}
  v16h afr[4];
  {
    const _Float16* rowp = &zn[hl][0];
    for (int kt = 0; kt < 4; ++kt) {
      int kb = kt * 32 + hi * 8;
      afr[kt] = frag_from(rowp + kb, rowp + kb + 16);
    }
  }

  v8f acc[5];
  for (int p = 0; p < 5; ++p) {
    v8f a = {};
    for (int kt = 0; kt < 4; ++kt) {
      const _Float16* wp = wsW + (size_t)p * 16384 + (size_t)(kt * 8 + wid) * 512
                         + lane * 16;
      v16h b = frag_from(wp, wp + 8);
      a = wmma_f16(afr[kt], b, a);
    }
    acc[p] = a;
  }

  const int n = wid * 16 + hl;
  const float vbq1 = bq1[n], vbq2 = bq2[n], vbk1 = bk1[n], vbk2 = bk2[n],
              vbv2 = bv2[n];
  for (int v = 0; v < 8; ++v) {
    int m   = v + hi * 8;
    int pos = posbase + m;
    float mk = smask[m];
    float q  = (acc[0][v] + vbq1) * sigf(acc[1][v] + vbq2) * mk;
    float kk = (acc[2][v] + vbk1) * sigf(acc[3][v] + vbk2) * mk;
    float vv = sigf(acc[4][v] + vbv2);
    int i = pos >> 9, j = pos & 511;
    wsQ[(size_t)n * NPOS + i * N_SEQ + j] = (_Float16)q;
    wsK[(size_t)n * NPOS + i * N_SEQ + j] = (_Float16)kk;
    wsV[(size_t)pos * C_DIM + n]          = (_Float16)vv;
  }
}

// ---------------------------------------------------------------------------
// Kernel 2: AT[q,k,c] = sum_r Q[q,r,c]*K[k,r,c].
// grid (4,4,128): one 128x128 output tile of channel c per block.
// Double-buffered async-to-LDS staging of 128x64 tiles (r-depth 64 = two
// WMMA k-steps per stage). Wave w computes a 64x32 sub-tile (4x2 fragments):
// 16 WMMAs per stage, 8 stages => 128 WMMAs / wave.
// ---------------------------------------------------------------------------
#define RDEPTH 64
#define NSTAGE (N_SEQ / RDEPTH)   // 8

__global__ __launch_bounds__(256)
void pass_b(const _Float16* __restrict__ wsQ, const _Float16* __restrict__ wsK,
            float* __restrict__ wsAT) {
  __shared__ __align__(16) _Float16 atile[2][128][RDEPTH];  // 2 x 16 KB
  __shared__ __align__(16) _Float16 btile[2][128][RDEPTH];  // 2 x 16 KB

  const int c  = blockIdx.z;
  const int qb = blockIdx.x * 128;
  const int kb = blockIdx.y * 128;
  const _Float16* Qp = wsQ + (size_t)c * NPOS;
  const _Float16* Kp = wsK + (size_t)c * NPOS;

  const int tid = threadIdx.x, wid = tid >> 5, lane = tid & 31;
  const int hl = lane & 15, hi = lane >> 4;
  const int wq = (wid >> 2) * 64;   // 0 / 64
  const int wk = (wid & 3) * 32;    // 0 / 32 / 64 / 96

  const int lrow = tid >> 1;          // 0..127
  const int seg  = (tid & 1) * 32;    // halves 0 / 32 (64 B per thread per tile)

  // Per stage each wave issues 8 async b128 ops (4 for A tile, 4 for B tile).
  auto issue_stage = [&](int s, int buf) {
    const _Float16* gq = Qp + (size_t)(qb + lrow) * N_SEQ + s * RDEPTH + seg;
    const _Float16* gk = Kp + (size_t)(kb + lrow) * N_SEQ + s * RDEPTH + seg;
    unsigned la = lds_off_of(&atile[buf][lrow][seg]);
    unsigned lb = lds_off_of(&btile[buf][lrow][seg]);
    for (int ch = 0; ch < 4; ++ch) {
      async_b128(la + ch * 16, gq + ch * 8);
      async_b128(lb + ch * 16, gk + ch * 8);
    }
  };

  v8f acc[4][2] = {};

  issue_stage(0, 0);
  for (int s = 0; s < NSTAGE; ++s) {
    const int buf = s & 1;
    if (s + 1 < NSTAGE) {
      issue_stage(s + 1, buf ^ 1);
      WAIT_ASYNC(8);     // 8 newest (next stage) may stay in flight; current done
    } else {
      WAIT_ASYNC(0);
    }
    __syncthreads();

    for (int kt = 0; kt < 2; ++kt) {
      v16h bf[2];
      for (int nn = 0; nn < 2; ++nn) {
        int col = wk + nn * 16 + hl;
        bf[nn] = frag_from(&btile[buf][col][kt * 32 + hi * 16],
                           &btile[buf][col][kt * 32 + hi * 16 + 8]);
      }
      for (int m = 0; m < 4; ++m) {
        int row = wq + m * 16 + hl;
        v16h af = frag_from(&atile[buf][row][kt * 32 + hi * 16],
                            &atile[buf][row][kt * 32 + hi * 16 + 8]);
        for (int nn = 0; nn < 2; ++nn)
          acc[m][nn] = wmma_f16(af, bf[nn], acc[m][nn]);
      }
    }
    __syncthreads();   // all waves done reading buf before it is overwritten
  }

  for (int m = 0; m < 4; ++m)
    for (int nn = 0; nn < 2; ++nn)
      for (int v = 0; v < 8; ++v) {
        int q = qb + wq + m * 16 + v + hi * 8;
        int k = kb + wk + nn * 16 + hl;
        wsAT[((size_t)q * N_SEQ + k) * C_DIM + c] = acc[m][nn][v];
      }
}

// ---------------------------------------------------------------------------
// Kernel 3: LN(AT) over channels + (ATn @ wv1 + bv1) * V -> f32 out.
// Same 16-position tiling as pass_a; wv1 is proj slot 5 in wsW.
// ---------------------------------------------------------------------------
__global__ __launch_bounds__(256)
void pass_c(const float* __restrict__ wsAT, const _Float16* __restrict__ wsV,
            const _Float16* __restrict__ wsW, const float* __restrict__ bv1,
            const float* __restrict__ g_out, const float* __restrict__ b_out,
            float* __restrict__ out) {
  __shared__ __align__(16) float    arow[16][C_DIM];
  __shared__ float    red1[16][16], red2[16][16];
  __shared__ __align__(16) _Float16 an[16][C_DIM];

  const int tid = threadIdx.x;
  const int posbase = blockIdx.x * 16;

  {
    const float* ga = wsAT + (size_t)posbase * C_DIM + tid * 8;
    unsigned l0 = lds_off_of(&arow[0][0] + tid * 8);
    async_b128(l0,      ga);
    async_b128(l0 + 16, ga + 4);
  }
  WAIT_ASYNC(0);
  __syncthreads();
  {
    int row = tid >> 4, l = tid & 15;
    float s1 = 0.f, s2 = 0.f;
    for (int e = 0; e < 8; ++e) {
      float x = arow[row][l * 8 + e];
      s1 += x; s2 += x * x;
    }
    red1[row][l] = s1; red2[row][l] = s2;
  }
  __syncthreads();
  {
    int row = tid >> 4, l = tid & 15;
    float s1 = 0.f, s2 = 0.f;
    for (int e = 0; e < 16; ++e) { s1 += red1[row][e]; s2 += red2[row][e]; }
    float mu  = s1 * (1.f / 128.f);
    float var = s2 * (1.f / 128.f) - mu * mu;
    float inv = rsqrtf(var + 1e-5f);
    for (int e = 0; e < 8; ++e) {
      int c = l * 8 + e;
      an[row][c] = (_Float16)((arow[row][c] - mu) * inv * g_out[c] + b_out[c]);
    }
  }
  __syncthreads();

  const int wid = tid >> 5, lane = tid & 31;
  const int hl = lane & 15, hi = lane >> 4;

  v16h afr[4];
  {
    const _Float16* rowp = &an[hl][0];
    for (int kt = 0; kt < 4; ++kt) {
      int kb = kt * 32 + hi * 8;
      afr[kt] = frag_from(rowp + kb, rowp + kb + 16);
    }
  }

  const _Float16* wbase = wsW + (size_t)5 * 16384;
  v8f acc = {};
  for (int kt = 0; kt < 4; ++kt) {
    const _Float16* wp = wbase + (size_t)(kt * 8 + wid) * 512 + lane * 16;
    v16h b = frag_from(wp, wp + 8);
    acc = wmma_f16(afr[kt], b, acc);
  }

  const int n = wid * 16 + hl;
  const float bias = bv1[n];
  for (int v = 0; v < 8; ++v) {
    int pos = posbase + v + hi * 8;
    float vv = (float)wsV[(size_t)pos * C_DIM + n];
    out[(size_t)pos * C_DIM + n] = (acc[v] + bias) * vv;
  }
}

// ---------------------------------------------------------------------------
extern "C" void kernel_launch(void* const* d_in, const int* in_sizes, int n_in,
                              void* d_out, int out_size, void* d_ws, size_t ws_size,
                              hipStream_t stream) {
  (void)in_sizes; (void)n_in; (void)out_size; (void)ws_size;
  const float* z    = (const float*)d_in[0];
  const float* mask = (const float*)d_in[1];
  const float* wq1  = (const float*)d_in[2];
  const float* bq1  = (const float*)d_in[3];
  const float* wq2  = (const float*)d_in[4];
  const float* bq2  = (const float*)d_in[5];
  const float* wk1  = (const float*)d_in[6];
  const float* bk1  = (const float*)d_in[7];
  const float* wk2  = (const float*)d_in[8];
  const float* bk2  = (const float*)d_in[9];
  const float* wv1  = (const float*)d_in[10];
  const float* bv1  = (const float*)d_in[11];
  const float* wv2  = (const float*)d_in[12];
  const float* bv2  = (const float*)d_in[13];
  const float* g_in = (const float*)d_in[14];
  const float* b_in = (const float*)d_in[15];
  const float* g_out= (const float*)d_in[16];
  const float* b_out= (const float*)d_in[17];
  float* out = (float*)d_out;

  char* ws = (char*)d_ws;
  _Float16* wsW = (_Float16*)ws;                 // 6 * 16384 halves = 192 KB
  size_t off = 6 * 16384 * sizeof(_Float16);
  _Float16* wsQ = (_Float16*)(ws + off); off += (size_t)C_DIM * NPOS * 2;  // 64 MB
  _Float16* wsK = (_Float16*)(ws + off); off += (size_t)C_DIM * NPOS * 2;  // 64 MB
  _Float16* wsV = (_Float16*)(ws + off); off += (size_t)C_DIM * NPOS * 2;  // 64 MB
  float*    wsAT = (float*)(ws + off);                                     // 128 MB

  prep_weights<<<(6 * C_DIM * C_DIM + 255) / 256, 256, 0, stream>>>(
      wq1, wq2, wk1, wk2, wv2, wv1, wsW);
  pass_a<<<NPOS / 16, 256, 0, stream>>>(z, mask, bq1, bq2, bk1, bk2, bv2,
                                        g_in, b_in, wsW, wsQ, wsK, wsV);
  pass_b<<<dim3(4, 4, C_DIM), 256, 0, stream>>>(wsQ, wsK, wsAT);
  pass_c<<<NPOS / 16, 256, 0, stream>>>(wsAT, wsV, wsW, bv1, g_out, b_out, out);
}
